// SkipBot_11922829214019
// MI455X (gfx1250) — matrix operand order repaired and checked by reading, not compiled, not run on hardware
//
#include <hip/hip_runtime.h>
#include <hip/hip_bf16.h>

// ---------------------------------------------------------------------------
// MI455X (gfx1250) transformer forward.
//   * all matmuls via v_wmma_f32_16x16x32_f16 (f16 in, f32 accum)
//   * GEMM A-tile staged with global_load_async_to_lds_b128 (ASYNCcnt)
//   * GEMM B-tile (weights, stored transposed [N][K]) staged with the
//     Tensor Data Mover: tensor_load_to_lds + s_wait_tensorcnt
// ---------------------------------------------------------------------------

typedef _Float16 v16h __attribute__((ext_vector_type(16)));
typedef _Float16 v8h  __attribute__((ext_vector_type(8)));
typedef float    v8f  __attribute__((ext_vector_type(8)));
typedef int      si4  __attribute__((ext_vector_type(4)));
typedef int      si8  __attribute__((ext_vector_type(8)));

#define B_    16
#define TCAT_ 512
#define TCH_  31
#define TSEQ  544                  // TCAT + 1 + TCH
#define C_    256
#define H_    8
#define HS_   32
#define L_    6
#define V_    512
#define NCONT_ 6
#define FFN_  1024
#define ROWS_ (B_ * TSEQ)          // 8704

// ------------------------- WMMA fragment helpers ---------------------------
// A (16x32 f16): lane l holds row m=l&15; elements K = kb+{0..7}, kb+16+{0..7},
// kb = 8*(l>>4).  Two 16B loads.
__device__ __forceinline__ v16h load_a_frag(const _Float16* p) {
  v8h lo = *(const v8h*)(p);
  v8h hi = *(const v8h*)(p + 16);
  v16h f;
#pragma unroll
  for (int i = 0; i < 8; ++i) { f[i] = lo[i]; f[i + 8] = hi[i]; }
  return f;
}
// B (32x16 f16): lane l holds col n=l&15; elements K = kb+{0..7}, kb+16+{0..7},
// kb = 8*(l>>4).  Two 16B loads.
__device__ __forceinline__ v16h load_b_frag(const _Float16* p) {
  v8h lo = *(const v8h*)(p);
  v8h hi = *(const v8h*)(p + 16);
  v16h f;
#pragma unroll
  for (int i = 0; i < 8; ++i) { f[i] = lo[i]; f[i + 8] = hi[i]; }
  return f;
}
__device__ __forceinline__ v8f wmma_f16(v16h a, v16h b, v8f c) {
  return __builtin_amdgcn_wmma_f32_16x16x32_f16(false, a, false, b,
                                                (short)0, c, false, false);
}

// ----------------------- CDNA5 async / TDM helpers -------------------------
// Per-lane async copy of 16B global -> LDS (GV mode).  Generic LDS pointer's
// low 32 bits are the wave-relative LDS byte offset (flat aperture rule).
__device__ __forceinline__ void async_g2l_b128(void* lds, const void* g) {
  unsigned l32 = (unsigned)(size_t)lds;
  unsigned long long ga = (unsigned long long)(size_t)g;
  asm volatile("global_load_async_to_lds_b128 %0, %1, off"
               :: "v"(l32), "v"(ga) : "memory");
}
__device__ __forceinline__ void wait_asynccnt0() {
  asm volatile("s_wait_asynccnt 0" ::: "memory");
}
// TDM: load a 32(k) x 64(n) f16 tile of a [N][K] row-major tensor into LDS
// (rows packed contiguously, 64B apart -> Bs[64][32]).  D# per ISA 08 §8.
__device__ __forceinline__ void tdm_load_b_tile(const _Float16* gsrc,
                                                void* ldsdst, int K, int N) {
  unsigned long long ga = (unsigned long long)(size_t)gsrc;
  si4 g0;
  g0[0] = 1;                                          // count=1, user D#
  g0[1] = (int)(unsigned)(size_t)ldsdst;              // lds_addr
  g0[2] = (int)(unsigned)(ga & 0xFFFFFFFFull);        // global_addr[31:0]
  g0[3] = (int)(((unsigned)(ga >> 32) & 0x01FFFFFFu)  // global_addr[56:32]
                | 0x80000000u);                       // type=2 ("image")
  si8 g1;
  g1[0] = 0x00010000;                                 // data_size=1 (2 bytes)
  g1[1] = (int)(((unsigned)K & 0xFFFFu) << 16);       // tensor_dim0 lo @63:48
  g1[2] = (int)(((unsigned)K >> 16) |                 // tensor_dim0 hi
                (((unsigned)N & 0xFFFFu) << 16));     // tensor_dim1 lo
  g1[3] = (int)(((unsigned)N >> 16) | (32u << 16));   // dim1 hi | tile_dim0=32
  g1[4] = 64;                                         // tile_dim1=64, dim2=0
  g1[5] = K;                                          // tensor_dim0_stride lo
  g1[6] = 0;                                          // stride0 hi, stride1 lo
  g1[7] = 0;
  asm volatile("tensor_load_to_lds %0, %1" :: "s"(g0), "s"(g1) : "memory");
}

// ------------------------------ embed --------------------------------------
__global__ __launch_bounds__(256) void embed_kernel(
    const int* __restrict__ cat, const float* __restrict__ cont,
    const int* __restrict__ choice, const float* __restrict__ state_emb,
    const float* __restrict__ cont_W, const float* __restrict__ cont_b,
    const float* __restrict__ choice_emb, float* __restrict__ x) {
  int t = blockIdx.x, b = blockIdx.y, c = threadIdx.x;
  float v;
  if (t < TCAT_) {
    v = state_emb[(size_t)cat[b * TCAT_ + t] * C_ + c];
  } else if (t == TCAT_) {
    v = cont_b[c];
#pragma unroll
    for (int j = 0; j < NCONT_; ++j)
      v += cont[b * NCONT_ + j] * cont_W[j * C_ + c];
  } else {
    v = choice_emb[(size_t)choice[b * TCH_ + (t - TCAT_ - 1)] * C_ + c];
  }
  x[((size_t)b * TSEQ + t) * C_ + c] = v;
}

// ---------------------------- layernorm ------------------------------------
__global__ __launch_bounds__(256) void ln_kernel(
    const float* __restrict__ x, const float* __restrict__ g,
    const float* __restrict__ b, _Float16* __restrict__ out) {
  int row = blockIdx.x, tid = threadIdx.x;
  float v = x[(size_t)row * C_ + tid];
  float s = v, s2 = v * v;
#pragma unroll
  for (int o = 16; o > 0; o >>= 1) {
    s  += __shfl_xor(s,  o, 32);
    s2 += __shfl_xor(s2, o, 32);
  }
  __shared__ float rs[8], rs2[8];
  int w = tid >> 5;
  if ((tid & 31) == 0) { rs[w] = s; rs2[w] = s2; }
  __syncthreads();
  float ts = 0.f, ts2 = 0.f;
#pragma unroll
  for (int i = 0; i < 8; ++i) { ts += rs[i]; ts2 += rs2[i]; }
  float mean = ts * (1.f / C_);
  float var  = ts2 * (1.f / C_) - mean * mean;
  float inv  = rsqrtf(var + 1e-5f);
  out[(size_t)row * C_ + tid] = (_Float16)((v - mean) * inv * g[tid] + b[tid]);
}

// ------------------------- weight conversion -------------------------------
// (L,H,C,HS) -> per-layer transposed [H*HS][C] f16 (so GEMM-B is [N][K]).
__global__ __launch_bounds__(256) void cvt_qkv_kernel(
    const float* __restrict__ in, _Float16* __restrict__ out) {
  int i = blockIdx.x * 256 + threadIdx.x;       // L*H*C*HS = 393216 total
  int d = i & (HS_ - 1);
  int c = (i >> 5) & (C_ - 1);
  int h = (i >> 13) & (H_ - 1);
  int l = i >> 16;
  out[(size_t)l * C_ * C_ + (size_t)(h * HS_ + d) * C_ + c] = (_Float16)in[i];
}
// per-layer [K][N] fp32 -> per-layer [N][K] f16
__global__ __launch_bounds__(256) void cvt_t_kernel(
    const float* __restrict__ in, _Float16* __restrict__ out,
    int Kd, int Nd, int total) {
  int i = blockIdx.x * 256 + threadIdx.x;
  if (i >= total) return;
  int per = Kd * Nd;
  int l = i / per, r = i - l * per;
  int k = r / Nd, n = r - k * Nd;
  out[(size_t)l * per + (size_t)n * Kd + k] = (_Float16)in[i];
}

// ------------------------------- GEMM --------------------------------------
// out[M,N] = A[M,K] (f16 row major) @ Bt[N,K]^T (f16, weights pre-transposed)
// flags: 1=+bias[n]  2=relu  4=+resid[m,n]  8=write f32  16=write f16
#define GF_BIAS 1
#define GF_RELU 2
#define GF_RES  4
#define GF_F32  8
#define GF_F16  16

__global__ __launch_bounds__(256) void gemm_kernel(
    const _Float16* __restrict__ A, const _Float16* __restrict__ Bt,
    const float* __restrict__ bias, const float* __restrict__ resid,
    float* __restrict__ outF, _Float16* __restrict__ outH,
    int M, int N, int K, int flags) {
  __shared__ __align__(16) _Float16 As[128][40];  // 80B row stride
  __shared__ __align__(16) _Float16 Bs[64][32];   // TDM tile, 64B row stride
  const int tid  = threadIdx.x;
  const int n0g  = blockIdx.x * 64;
  const int m0   = blockIdx.y * 128;
  const int w    = tid >> 5, lane = tid & 31;
  const int wm   = (w >> 1) * 32, wn = (w & 1) * 32;
  const int lm   = lane & 15;
  const int ka   = (lane >> 4) * 8;     // A frag K base
  const int kb2  = (lane >> 4) * 8;     // B frag K base
  const int wid  = __builtin_amdgcn_readfirstlane(tid >> 5);  // scalar wave id

  v8f acc[2][2] = {};

  const int ar = tid >> 1, ac = (tid & 1) * 16;   // A stage: 128 rows x 32
  int gr = m0 + ar; if (gr >= M) gr = M - 1;      // clamp (stores are guarded)

  for (int k0 = 0; k0 < K; k0 += 32) {
    const _Float16* pa = A + (size_t)gr * K + k0 + ac;
    __syncthreads();                      // previous tile fully consumed
    async_g2l_b128(&As[ar][ac],     pa);
    async_g2l_b128(&As[ar][ac + 8], pa + 8);
    if (wid == 0)
      tdm_load_b_tile(Bt + (size_t)n0g * K + k0, &Bs[0][0], K, N);
    if (k0 + 32 < K) __builtin_prefetch(pa + 32, 0, 0);
    wait_asynccnt0();
    if (wid == 0) __builtin_amdgcn_s_wait_tensorcnt(0);
    __syncthreads();

    v16h af[2], bf[2];
#pragma unroll
    for (int i = 0; i < 2; ++i) af[i] = load_a_frag(&As[wm + i * 16 + lm][ka]);
#pragma unroll
    for (int j = 0; j < 2; ++j) bf[j] = load_b_frag(&Bs[wn + j * 16 + lm][kb2]);
#pragma unroll
    for (int i = 0; i < 2; ++i)
#pragma unroll
      for (int j = 0; j < 2; ++j)
        acc[i][j] = wmma_f16(af[i], bf[j], acc[i][j]);
  }

  const bool hasB = flags & GF_BIAS, doRelu = flags & GF_RELU;
  const bool hasR = flags & GF_RES,  wF = flags & GF_F32, wH = flags & GF_F16;
#pragma unroll
  for (int i = 0; i < 2; ++i) {
#pragma unroll
    for (int j = 0; j < 2; ++j) {
      const int n = n0g + wn + j * 16 + lm;
      const float bb = hasB ? bias[n] : 0.f;
#pragma unroll
      for (int r = 0; r < 8; ++r) {
        const int m = m0 + wm + i * 16 + r + ((lane >> 4) << 3);
        if (m < M) {
          float vv = acc[i][j][r] + bb;
          if (doRelu) vv = fmaxf(vv, 0.f);
          const size_t idx = (size_t)m * N + n;
          if (hasR) vv += resid[idx];
          if (wF) outF[idx] = vv;
          if (wH) outH[idx] = (_Float16)vv;
        }
      }
    }
  }
}

// ----------------------------- attention -----------------------------------
// One wave per (qtile, head, batch). scores in LDS, softmax, W@V via WMMA.
__global__ __launch_bounds__(32) void attn_kernel(
    const _Float16* __restrict__ q, const _Float16* __restrict__ k,
    const _Float16* __restrict__ v, _Float16* __restrict__ o) {
  __shared__ __align__(16) float    sc[16][TSEQ];   // 34816 B
  __shared__ __align__(16) _Float16 wl[16][TSEQ];   // 17408 B
  __shared__ __align__(16) _Float16 vs[32][32];     //  2048 B
  const int qt = blockIdx.x, h = blockIdx.y, b = blockIdx.z;
  const int l  = threadIdx.x;
  const int lm = l & 15;
  const int ka = (l >> 4) * 8;
  const int kb2 = (l >> 4) * 8;
  const float scale = 0.0625f;   // C^-0.5 = 1/16 (full-embed scaling)

  // Q fragment (A): row t = qt*16+lm, K = head dims (contiguous).
  const v16h af = load_a_frag(
      q + ((size_t)(b * TSEQ + qt * 16 + lm)) * C_ + h * HS_ + ka);

  // scores = Q K^T * scale  (34 key tiles of 16)
  for (int st = 0; st < TSEQ / 16; ++st) {
    const v16h bf = load_b_frag(
        k + ((size_t)(b * TSEQ + st * 16 + lm)) * C_ + h * HS_ + kb2);
    v8f cm = {};
    cm = wmma_f16(af, bf, cm);
#pragma unroll
    for (int r = 0; r < 8; ++r)
      sc[r + ((l >> 4) << 3)][st * 16 + lm] = cm[r] * scale;
  }
  __syncthreads();

  // softmax over full row: lane pair (lm, lm+16) split the 544 columns
  {
    const int c0 = (l >> 4) * (TSEQ / 2);
    float mx = -1e30f;
    for (int s = 0; s < TSEQ / 2; ++s) mx = fmaxf(mx, sc[lm][c0 + s]);
    mx = fmaxf(mx, __shfl_xor(mx, 16, 32));
    float sum = 0.f;
    for (int s = 0; s < TSEQ / 2; ++s) {
      float e = __expf(sc[lm][c0 + s] - mx);
      sum += e;
      sc[lm][c0 + s] = e;
    }
    sum += __shfl_xor(sum, 16, 32);
    const float inv = 1.f / sum;
    for (int s = 0; s < TSEQ / 2; ++s)
      wl[lm][c0 + s] = (_Float16)(sc[lm][c0 + s] * inv);
  }
  __syncthreads();

  // O = W @ V   (K = T = 544 = 17 chunks of 32)
  v8f acc0 = {}, acc1 = {};
  for (int kc = 0; kc < TSEQ / 32; ++kc) {
    const _Float16* pv = v + ((size_t)(b * TSEQ + kc * 32 + l)) * C_ + h * HS_;
    v8h r0 = *(const v8h*)(pv);
    v8h r1 = *(const v8h*)(pv + 8);
    v8h r2 = *(const v8h*)(pv + 16);
    v8h r3 = *(const v8h*)(pv + 24);
    __syncthreads();                     // prior chunk consumed
#pragma unroll
    for (int d = 0; d < 8; ++d) {
      vs[d][l]      = r0[d];
      vs[d + 8][l]  = r1[d];
      vs[d + 16][l] = r2[d];
      vs[d + 24][l] = r3[d];
    }
    __syncthreads();
    const v16h wa = load_a_frag(&wl[lm][kc * 32 + ka]);
    const v16h b0 = load_b_frag(&vs[lm][kb2]);
    const v16h b1 = load_b_frag(&vs[16 + lm][kb2]);
    acc0 = wmma_f16(wa, b0, acc0);
    acc1 = wmma_f16(wa, b1, acc1);
  }

#pragma unroll
  for (int r = 0; r < 8; ++r) {
    const int t = qt * 16 + r + ((l >> 4) << 3);
    const size_t base = ((size_t)(b * TSEQ + t)) * C_ + h * HS_;
    o[base + lm]      = (_Float16)acc0[r];
    o[base + 16 + lm] = (_Float16)acc1[r];
  }
}

// ------------------------------- pool --------------------------------------
__global__ __launch_bounds__(256) void pool_kernel(
    const _Float16* __restrict__ hx, _Float16* __restrict__ pooled) {
  int b = blockIdx.x, c = threadIdx.x;
  float s = 0.f;
  for (int t = 0; t < TSEQ; ++t)
    s += (float)hx[((size_t)b * TSEQ + t) * C_ + c];
  pooled[b * C_ + c] = (_Float16)s;
}

// ----------------------------- launcher ------------------------------------
extern "C" void kernel_launch(void* const* d_in, const int* in_sizes, int n_in,
                              void* d_out, int out_size, void* d_ws,
                              size_t ws_size, hipStream_t stream) {
  (void)in_sizes; (void)n_in; (void)out_size; (void)ws_size;
  const int*   cat        = (const int*)  d_in[0];
  const float* cont       = (const float*)d_in[1];
  const int*   choice     = (const int*)  d_in[2];
  const float* state_emb  = (const float*)d_in[3];
  const float* cont_W     = (const float*)d_in[4];
  const float* cont_b     = (const float*)d_in[5];
  const float* choice_emb = (const float*)d_in[6];
  const float* Wq         = (const float*)d_in[7];
  const float* Wk         = (const float*)d_in[8];
  const float* Wv         = (const float*)d_in[9];
  const float* proj_W     = (const float*)d_in[10];
  const float* proj_b     = (const float*)d_in[11];
  const float* ln1_g      = (const float*)d_in[12];
  const float* ln1_b      = (const float*)d_in[13];
  const float* ln2_g      = (const float*)d_in[14];
  const float* ln2_b      = (const float*)d_in[15];
  const float* ffn_W1     = (const float*)d_in[16];
  const float* ffn_b1     = (const float*)d_in[17];
  const float* ffn_W2     = (const float*)d_in[18];
  const float* ffn_b2     = (const float*)d_in[19];
  const float* lnf_g      = (const float*)d_in[20];
  const float* lnf_b      = (const float*)d_in[21];
  const float* head_W     = (const float*)d_in[22];
  const float* head_b     = (const float*)d_in[23];
  float* out = (float*)d_out;

  // workspace partition (256B aligned)
  size_t off = 0;
  auto wsa = [&](size_t bytes) -> void* {
    void* p = (char*)d_ws + off;
    off += (bytes + 255) & ~(size_t)255;
    return p;
  };
  float*    x       = (float*)   wsa((size_t)ROWS_ * C_ * 4);
  _Float16* hbuf    = (_Float16*)wsa((size_t)ROWS_ * C_ * 2);
  _Float16* q16     = (_Float16*)wsa((size_t)ROWS_ * C_ * 2);
  _Float16* k16     = (_Float16*)wsa((size_t)ROWS_ * C_ * 2);
  _Float16* v16     = (_Float16*)wsa((size_t)ROWS_ * C_ * 2);
  _Float16* o16     = (_Float16*)wsa((size_t)ROWS_ * C_ * 2);
  _Float16* u16     = (_Float16*)wsa((size_t)ROWS_ * FFN_ * 2);
  _Float16* wq16    = (_Float16*)wsa((size_t)L_ * C_ * C_ * 2);
  _Float16* wk16    = (_Float16*)wsa((size_t)L_ * C_ * C_ * 2);
  _Float16* wv16    = (_Float16*)wsa((size_t)L_ * C_ * C_ * 2);
  _Float16* wp16    = (_Float16*)wsa((size_t)L_ * C_ * C_ * 2);
  _Float16* w116    = (_Float16*)wsa((size_t)L_ * C_ * FFN_ * 2);
  _Float16* w216    = (_Float16*)wsa((size_t)L_ * FFN_ * C_ * 2);
  _Float16* wh16    = (_Float16*)wsa((size_t)C_ * V_ * 2);
  _Float16* pooled  = (_Float16*)wsa((size_t)B_ * C_ * 2);

  // weight conversion / transposition to f16 [N][K]
  {
    const int nqkv = L_ * H_ * C_ * HS_;                        // 393216
    cvt_qkv_kernel<<<nqkv / 256, 256, 0, stream>>>(Wq, wq16);
    cvt_qkv_kernel<<<nqkv / 256, 256, 0, stream>>>(Wk, wk16);
    cvt_qkv_kernel<<<nqkv / 256, 256, 0, stream>>>(Wv, wv16);
    const int npj = L_ * C_ * C_;
    cvt_t_kernel<<<(npj + 255) / 256, 256, 0, stream>>>(proj_W, wp16, C_, C_, npj);
    const int nf = L_ * C_ * FFN_;
    cvt_t_kernel<<<(nf + 255) / 256, 256, 0, stream>>>(ffn_W1, w116, C_, FFN_, nf);
    cvt_t_kernel<<<(nf + 255) / 256, 256, 0, stream>>>(ffn_W2, w216, FFN_, C_, nf);
    const int nh = C_ * V_;
    cvt_t_kernel<<<(nh + 255) / 256, 256, 0, stream>>>(head_W, wh16, C_, V_, nh);
  }

  // embedding concat
  embed_kernel<<<dim3(TSEQ, B_), 256, 0, stream>>>(
      cat, cont, choice, state_emb, cont_W, cont_b, choice_emb, x);

  const dim3 gQKV(C_ / 64, ROWS_ / 128);
  const dim3 gF1(FFN_ / 64, ROWS_ / 128);
  for (int l = 0; l < L_; ++l) {
    // ln1 -> h (f16)
    ln_kernel<<<ROWS_, 256, 0, stream>>>(x, ln1_g + l * C_, ln1_b + l * C_, hbuf);
    // q, k, v
    gemm_kernel<<<gQKV, 256, 0, stream>>>(hbuf, wq16 + (size_t)l * C_ * C_,
        nullptr, nullptr, nullptr, q16, ROWS_, C_, C_, GF_F16);
    gemm_kernel<<<gQKV, 256, 0, stream>>>(hbuf, wk16 + (size_t)l * C_ * C_,
        nullptr, nullptr, nullptr, k16, ROWS_, C_, C_, GF_F16);
    gemm_kernel<<<gQKV, 256, 0, stream>>>(hbuf, wv16 + (size_t)l * C_ * C_,
        nullptr, nullptr, nullptr, v16, ROWS_, C_, C_, GF_F16);
    // attention
    attn_kernel<<<dim3(TSEQ / 16, H_, B_), 32, 0, stream>>>(q16, k16, v16, o16);
    // x += o @ proj_W + proj_b
    gemm_kernel<<<gQKV, 256, 0, stream>>>(o16, wp16 + (size_t)l * C_ * C_,
        proj_b + l * C_, x, x, nullptr, ROWS_, C_, C_, GF_BIAS | GF_RES | GF_F32);
    // ln2 -> h (f16)
    ln_kernel<<<ROWS_, 256, 0, stream>>>(x, ln2_g + l * C_, ln2_b + l * C_, hbuf);
    // u = relu(h @ W1 + b1)
    gemm_kernel<<<gF1, 256, 0, stream>>>(hbuf, w116 + (size_t)l * C_ * FFN_,
        ffn_b1 + l * FFN_, nullptr, nullptr, u16, ROWS_, FFN_, C_,
        GF_BIAS | GF_RELU | GF_F16);
    // x += u @ W2 + b2
    gemm_kernel<<<gQKV, 256, 0, stream>>>(u16, w216 + (size_t)l * FFN_ * C_,
        ffn_b2 + l * C_, x, x, nullptr, ROWS_, C_, FFN_,
        GF_BIAS | GF_RES | GF_F32);
  }

  // final LN -> f16, sum-pool, head
  ln_kernel<<<ROWS_, 256, 0, stream>>>(x, lnf_g, lnf_b, hbuf);
  pool_kernel<<<B_, 256, 0, stream>>>(hbuf, pooled);
  gemm_kernel<<<dim3(V_ / 64, 1), 256, 0, stream>>>(pooled, wh16, head_b,
      nullptr, out, nullptr, B_, V_, C_, GF_BIAS | GF_F32);
}
